// PolicyNetworkID_82411832475890
// MI455X (gfx1250) — matrix operand order repaired and checked by reading, not compiled, not run on hardware
//
#include <hip/hip_runtime.h>
#include <hip/hip_bf16.h>

#define RHO   1.0f
#define SIGMA 1e-6f

typedef __attribute__((ext_vector_type(16))) __bf16 v16bf;
typedef __attribute__((ext_vector_type(8)))  float  v8f;

__device__ __forceinline__ unsigned short f2bf(float f) {
    unsigned int u = __float_as_uint(f);
    unsigned int r = u + 0x7fffu + ((u >> 16) & 1u);   // round-to-nearest-even
    return (unsigned short)(r >> 16);
}

// ---------------------------------------------------------------- conversions
__global__ void cvt_bf16_kernel(const float* __restrict__ src,
                                unsigned short* __restrict__ dst, int n) {
    int i = blockIdx.x * blockDim.x + threadIdx.x;
    if (i < n) dst[i] = f2bf(src[i]);
}

// [K][N] f32 -> [Npad][K] bf16 (transposed, zero-padded cols)
__global__ void cvt_transpose_kernel(const float* __restrict__ src,
                                     unsigned short* __restrict__ dst,
                                     int K, int N, int Npad) {
    int i = blockIdx.x * blockDim.x + threadIdx.x;
    if (i < Npad * K) {
        int n = i / K, k = i % K;
        dst[i] = (n < N) ? f2bf(src[(size_t)k * N + n]) : (unsigned short)0;
    }
}

// ---------------------------------------------------------------- QP setup
__global__ void qp_setup_kernel(const float* __restrict__ Aeq,
                                const float* __restrict__ beq,
                                const float* __restrict__ Ain,
                                const float* __restrict__ bin,
                                const float* __restrict__ ub,
                                const float* __restrict__ lb,
                                float* __restrict__ Atot,
                                float* __restrict__ lt,
                                float* __restrict__ ut,
                                float* __restrict__ Minv) {
    __shared__ float sA[60 * 24];
    __shared__ float aug[24 * 48];
    __shared__ float fac[24];
    const int tid = threadIdx.x, nth = blockDim.x;

    for (int idx = tid; idx < 60 * 24; idx += nth) {
        int m = idx / 24, j = idx % 24;
        float v;
        if      (m < 4)  v = Aeq[m * 24 + j];
        else if (m < 12) v = Ain[(m - 4) * 24 + j];
        else if (m < 36) v = (m - 12 == j) ?  1.f : 0.f;
        else             v = (m - 36 == j) ? -1.f : 0.f;
        sA[idx] = v; Atot[idx] = v;
    }
    for (int m = tid; m < 60; m += nth) {
        float lo, hi;
        if      (m < 4)  { lo = beq[m]; hi = beq[m]; }
        else if (m < 12) { lo = -__builtin_inff(); hi = bin[m - 4]; }
        else if (m < 36) { int j = m - 12; lo =  lb[j]; hi =  ub[j]; }
        else             { int j = m - 36; lo = -ub[j]; hi = -lb[j]; }
        lt[m] = lo; ut[m] = hi;
    }
    __syncthreads();

    for (int idx = tid; idx < 24 * 48; idx += nth) {
        int r = idx / 48, c = idx % 48;
        float v;
        if (c < 24) {
            float s = 0.f;
            for (int m = 0; m < 60; m++) s += sA[m * 24 + r] * sA[m * 24 + c];
            v = (r == c ? (1.f + SIGMA) : 0.f) + RHO * s;
        } else {
            v = (c - 24 == r) ? 1.f : 0.f;
        }
        aug[idx] = v;
    }
    __syncthreads();

    for (int k = 0; k < 24; k++) {
        float piv = aug[k * 48 + k];
        __syncthreads();
        for (int c = tid; c < 48; c += nth) aug[k * 48 + c] /= piv;
        __syncthreads();
        if (tid < 24) fac[tid] = aug[tid * 48 + k];
        __syncthreads();
        for (int idx = tid; idx < 24 * 48; idx += nth) {
            int r = idx / 48, c = idx % 48;
            if (r != k) aug[idx] -= fac[r] * aug[k * 48 + c];
        }
        __syncthreads();
    }
    for (int idx = tid; idx < 24 * 24; idx += nth) {
        int r = idx / 24, c = idx % 24;
        Minv[idx] = aug[r * 48 + 24 + c];
    }
}

// ---------------------------------------------------------------- WMMA GEMM
// C[M,Nout] = act(A[M,K](bf16, row-major) @ Bt[Nld][K](bf16, col-major) + bias)
// Block tile 128x128, 8 waves (4x2), each wave 32x64 = 2x4 wmma 16x16x32 tiles.
// All LDS fragment traffic is explicit ds_load_b128.
#define LDSTRIDE 40   // 32 K-elems + 8 pad ushorts (80B) to spread banks

union Frag { v16bf v; uint4 q[2]; };
union U8   { uint4 v; unsigned short s[8]; };

template <bool RELU, bool OUT_BF16, bool NGUARD>
__global__ __launch_bounds__(256) void gemm_bf16_wmma(
        const unsigned short* __restrict__ Ap,   // [M][K]
        const unsigned short* __restrict__ Bt,   // [Nld][K]
        const float* __restrict__ bias,
        void* __restrict__ Cout,
        int M, int K, int Nld, int Nout, int ldOut) {
    __shared__ unsigned short lds_a[128 * LDSTRIDE];   // row-major rows, K inner
    __shared__ unsigned short lds_b[128 * LDSTRIDE];   // col-major cols, K inner

    const int tid     = threadIdx.x;
    const int lane    = tid & 31;
    const int wave    = tid >> 5;
    const int waveRow = wave >> 1;   // 0..3  -> 32 rows each
    const int waveCol = wave & 1;    // 0..1  -> 64 cols each
    const int half    = lane >> 4;
    const int l15     = lane & 15;

    const int rowBase = blockIdx.y * 128;
    const int colBase = blockIdx.x * 128;

    const v8f zero = {0.f,0.f,0.f,0.f,0.f,0.f,0.f,0.f};
    v8f acc[2][4];
    #pragma unroll
    for (int i = 0; i < 2; i++)
        #pragma unroll
        for (int j = 0; j < 4; j++) acc[i][j] = zero;

    // staging indices: A tile 128x32 (16 ushorts/thread), B tile 128x32 (16/thread)
    const int arow = tid >> 1, acol = (tid & 1) * 16;
    const int bcolT = tid >> 1, bkoff = (tid & 1) * 16;

    for (int kk = 0; kk < K; kk += 32) {
        __syncthreads();
        // --- stage A tile
        {
            const unsigned short* gA = Ap + (size_t)(rowBase + arow) * K + kk + acol;
            *(uint4*)&lds_a[arow * LDSTRIDE + acol]     = *(const uint4*)gA;
            *(uint4*)&lds_a[arow * LDSTRIDE + acol + 8] = *(const uint4*)(gA + 8);
        }
        // --- stage B tile (column-major, contiguous K runs)
        {
            const int gcol = colBase + bcolT;
            uint4 b0, b1;
            if (!NGUARD || gcol < Nld) {
                const unsigned short* gB = Bt + (size_t)gcol * K + kk + bkoff;
                b0 = *(const uint4*)gB;
                b1 = *(const uint4*)(gB + 8);
            } else {
                b0 = make_uint4(0u, 0u, 0u, 0u);
                b1 = b0;
            }
            *(uint4*)&lds_b[bcolT * LDSTRIDE + bkoff]     = b0;
            *(uint4*)&lds_b[bcolT * LDSTRIDE + bkoff + 8] = b1;
        }
        __syncthreads();

        // --- fragments: two ds_load_b128 each (contiguous K = 16h + s, same order in A and B)
        Frag a[2];
        #pragma unroll
        for (int i = 0; i < 2; i++) {
            const unsigned short* ap =
                &lds_a[(waveRow * 32 + i * 16 + l15) * LDSTRIDE];
            a[i].q[0] = *(const uint4*)&ap[16 * half];       // K = 16h .. 16h+7
            a[i].q[1] = *(const uint4*)&ap[16 * half + 8];   // K = 16h+8 .. +15
        }
        Frag b[4];
        #pragma unroll
        for (int j = 0; j < 4; j++) {
            const unsigned short* bp =
                &lds_b[(waveCol * 64 + j * 16 + l15) * LDSTRIDE + 16 * half];
            b[j].q[0] = *(const uint4*)&bp[0];               // K = 16h .. 16h+7
            b[j].q[1] = *(const uint4*)&bp[8];               // K = 16h+8 .. +15
        }
        #pragma unroll
        for (int i = 0; i < 2; i++)
            #pragma unroll
            for (int j = 0; j < 4; j++)
                acc[i][j] = __builtin_amdgcn_wmma_f32_16x16x32_bf16(
                    false, a[i].v, false, b[j].v, short(0), acc[i][j], false, false);
    }

    // --- epilogue: bias + optional relu, bf16 or f32 store (C/D layout)
    unsigned short* outB = (unsigned short*)Cout;
    float*          outF = (float*)Cout;
    #pragma unroll
    for (int j = 0; j < 4; j++) {
        const int colg = colBase + waveCol * 64 + j * 16 + l15;
        if (NGUARD && colg >= Nout) continue;
        const float bb = bias[colg];
        #pragma unroll
        for (int i = 0; i < 2; i++) {
            #pragma unroll
            for (int vg = 0; vg < 8; vg++) {
                const int rowg = rowBase + waveRow * 32 + i * 16 + half * 8 + vg;
                float val = acc[i][j][vg] + bb;
                if (RELU) val = fmaxf(val, 0.f);
                if (OUT_BF16) outB[(size_t)rowg * ldOut + colg] = f2bf(val);
                else          outF[(size_t)rowg * ldOut + colg] = val;
            }
        }
    }
}

// ---------------------------------------------------------------- ADMM solve
__global__ __launch_bounds__(256) void admm_kernel(
        const float* __restrict__ raw,
        const float* __restrict__ Atot,
        const float* __restrict__ Minv,
        const float* __restrict__ lt,
        const float* __restrict__ ut,
        float* __restrict__ Xout) {
    __shared__ float sA[60 * 24];
    __shared__ float sM[24 * 24];
    __shared__ float sl[60], su[60];
    const int tid = threadIdx.x;
    for (int i = tid; i < 60 * 24; i += 256) sA[i] = Atot[i];
    for (int i = tid; i < 24 * 24; i += 256) sM[i] = Minv[i];
    if (tid < 60) { sl[tid] = lt[tid]; su[tid] = ut[tid]; }
    __syncthreads();

    const int row = blockIdx.x * 256 + tid;
    float q[24], X[24], Z[60], Y[60];
    #pragma unroll
    for (int j = 0; j < 24; j++) { q[j] = -raw[(size_t)row * 24 + j]; X[j] = 0.f; }
    #pragma unroll
    for (int m = 0; m < 60; m++) { Z[m] = 0.f; Y[m] = 0.f; }

    for (int it = 0; it < 100; ++it) {
        float rhs[24];
        #pragma unroll
        for (int j = 0; j < 24; j++) rhs[j] = SIGMA * X[j] - q[j];
        #pragma unroll
        for (int m = 0; m < 60; m++) {
            const float w = RHO * Z[m] - Y[m];
            const float* ar = &sA[m * 24];
            #pragma unroll
            for (int j = 0; j < 24; j++) rhs[j] = fmaf(w, ar[j], rhs[j]);
        }
        #pragma unroll
        for (int i = 0; i < 24; i++) {
            const float* mr = &sM[i * 24];
            float s = 0.f;
            #pragma unroll
            for (int j = 0; j < 24; j++) s = fmaf(rhs[j], mr[j], s);
            X[i] = s;
        }
        #pragma unroll
        for (int m = 0; m < 60; m++) {
            const float* ar = &sA[m * 24];
            float s = 0.f;
            #pragma unroll
            for (int j = 0; j < 24; j++) s = fmaf(X[j], ar[j], s);
            const float zt = s;
            const float zn = fminf(fmaxf(zt + Y[m] / RHO, sl[m]), su[m]);
            Y[m] = Y[m] + RHO * (zt - zn);
            Z[m] = zn;
        }
    }
    #pragma unroll
    for (int j = 0; j < 24; j++) Xout[(size_t)row * 24 + j] = X[j];
}

// ---------------------------------------------------------------- launch
extern "C" void kernel_launch(void* const* d_in, const int* in_sizes, int n_in,
                              void* d_out, int out_size, void* d_ws, size_t ws_size,
                              hipStream_t stream) {
    (void)in_sizes; (void)n_in; (void)out_size; (void)ws_size;
    const int B = 16384, S = 512, H = 1024;

    const float* state = (const float*)d_in[0];
    const float* Aeq   = (const float*)d_in[1];
    const float* beq   = (const float*)d_in[2];
    const float* Ain   = (const float*)d_in[3];
    const float* bin   = (const float*)d_in[4];
    const float* ub    = (const float*)d_in[5];
    const float* lb    = (const float*)d_in[6];
    const float* W1    = (const float*)d_in[7];
    const float* b1    = (const float*)d_in[8];
    const float* W2    = (const float*)d_in[9];
    const float* b2    = (const float*)d_in[10];
    const float* W3    = (const float*)d_in[11];
    const float* b3    = (const float*)d_in[12];

    char*  ws = (char*)d_ws;
    size_t o  = 0;
    auto take = [&](size_t bytes) -> void* {
        void* p = ws + o;
        o += (bytes + 255) & ~size_t(255);
        return p;
    };
    unsigned short* state_bf = (unsigned short*)take((size_t)B * S * 2);
    unsigned short* w1t      = (unsigned short*)take((size_t)H * S * 2);   // [H][S]
    unsigned short* w2t      = (unsigned short*)take((size_t)H * H * 2);   // [H][H]
    unsigned short* w3t      = (unsigned short*)take((size_t)32 * H * 2);  // [32][H]
    unsigned short* h1       = (unsigned short*)take((size_t)B * H * 2);
    unsigned short* h2       = (unsigned short*)take((size_t)B * H * 2);
    float*          raw      = (float*)take((size_t)B * 24 * 4);
    float*          Atot     = (float*)take(60 * 24 * 4);
    float*          lt       = (float*)take(60 * 4);
    float*          ut       = (float*)take(60 * 4);
    float*          Minv     = (float*)take(24 * 24 * 4);

    // 1) precision staging (+ one-time weight transposes to col-major)
    cvt_bf16_kernel<<<(B * S + 255) / 256, 256, 0, stream>>>(state, state_bf, B * S);
    cvt_transpose_kernel<<<(H * S + 255) / 256, 256, 0, stream>>>(W1, w1t, S, H, H);
    cvt_transpose_kernel<<<(H * H + 255) / 256, 256, 0, stream>>>(W2, w2t, H, H, H);
    cvt_transpose_kernel<<<(32 * H + 255) / 256, 256, 0, stream>>>(W3, w3t, H, 24, 32);

    // 2) QP constants
    qp_setup_kernel<<<1, 64, 0, stream>>>(Aeq, beq, Ain, bin, ub, lb, Atot, lt, ut, Minv);

    // 3) MLP on the WMMA path (block tile 128x128)
    gemm_bf16_wmma<true,  true,  false><<<dim3(H / 128, B / 128), 256, 0, stream>>>(
        state_bf, w1t, b1, h1, B, S, H, H, H);
    gemm_bf16_wmma<true,  true,  false><<<dim3(H / 128, B / 128), 256, 0, stream>>>(
        h1, w2t, b2, h2, B, H, H, H, H);
    gemm_bf16_wmma<false, false, true ><<<dim3(1, B / 128), 256, 0, stream>>>(
        h2, w3t, b3, raw, B, H, 32, 24, 24);

    // 4) ADMM
    admm_kernel<<<B / 256, 256, 0, stream>>>(raw, Atot, Minv, lt, ut, (float*)d_out);
}
